// DLinearLayer_node_60610578481277
// MI455X (gfx1250) — compile-verified, matrix operand order checked by reading, not run on hardware
//
#include <hip/hip_runtime.h>
#include <hip/hip_bf16.h>

// ---------------------------------------------------------------------------
// DLinear layer for MI455X (gfx1250): per-feature dual GEMM
//   out[b,n,f] = sum_t trend[b,t,f]*TW[f,t,n] + Tb[f,n]
//              + sum_t resid[b,t,f]*RW[f,t,n] + Rb[f,n]
// bf16 operands + f32 accumulation via v_wmma_f32_16x16x32_bf16 (wave32).
// Per workgroup: full B=256 rows x 64 N-columns for one feature -> each
// weight element streams from HBM exactly once (23.3 TB/s bound ~7us).
// 180 KB double-buffered LDS pipeline (CDNA5 320 KB/WGP).
// ---------------------------------------------------------------------------

typedef __bf16 bf16;
typedef __attribute__((ext_vector_type(2)))  __bf16 bf16x2;
typedef __attribute__((ext_vector_type(8)))  __bf16 bf16x8;
typedef __attribute__((ext_vector_type(16))) __bf16 bf16x16;
typedef __attribute__((ext_vector_type(4)))  float  f32x4;
typedef __attribute__((ext_vector_type(8)))  float  f32x8;

#define B_  256
#define T_  1024
#define F_  16
#define N_  1024
#define KT  64                    // K (time) chunk per pipeline stage
#define NT  64                    // N columns per workgroup
#define SK  (KT + 8)              // padded LDS row stride (elements, 144 B)
#define NC  (T_ / KT)             // 16 K-chunks

#define A_STAGE (B_ * SK)         // elements per A stage (per matrix)
#define W_STAGE (NT * SK)         // elements per W stage (per matrix)
#define SMEM_BYTES ((size_t)(2 * 2 * A_STAGE + 2 * 2 * W_STAGE) * sizeof(__bf16))

__global__ __launch_bounds__(256) void dlinear_wmma_kernel(
    const float* __restrict__ hist,   // [B,T,F]
    const float* __restrict__ TW,     // [F,T,N]
    const float* __restrict__ Tb,     // [F,N]
    const float* __restrict__ RW,     // [F,T,N]
    const float* __restrict__ Rb,     // [F,N]
    float* __restrict__ out)          // [B,N,F]
{
  extern __shared__ __align__(16) char smem_raw[];
  bf16* sAt = (bf16*)smem_raw;              // [2][B_][SK]  trend A
  bf16* sAr = sAt + 2 * A_STAGE;            // [2][B_][SK]  residual A
  bf16* sWt = sAr + 2 * A_STAGE;            // [2][NT][SK]  trend W (transposed [n][k])
  bf16* sWr = sWt + 2 * W_STAGE;            // [2][NT][SK]  residual W

  const int tid  = threadIdx.x;
  const int f    = blockIdx.y;
  const int n0   = blockIdx.x * NT;

  const int wave = tid >> 5;
  const int lane = tid & 31;
  const int l16  = lane & 15;
  const int hs   = lane >> 4;     // K half-select per WMMA 16-bit operand layout
  const int wm   = wave & 3;      // 4 row-waves * 64 rows
  const int wn   = wave >> 2;     // 2 col-waves * 32 cols

  f32x8 acc[4][2];
  #pragma unroll
  for (int i = 0; i < 4; ++i)
    #pragma unroll
    for (int j = 0; j < 2; ++j) acc[i][j] = (f32x8)0.0f;

  // -------- stage loader: global -> (trend/resid transform, bf16) -> LDS ----
  auto load_stage = [&](int kc, int s) {
    const int k0 = kc * KT;
    bf16* at = sAt + s * A_STAGE;
    bf16* ar = sAr + s * A_STAGE;
    bf16* wt = sWt + s * W_STAGE;
    bf16* wr = sWr + s * W_STAGE;

    // A: one history row (b = tid) per thread; derive trend/residual.
    {
      const float* src = hist + (size_t)tid * (T_ * F_) + (size_t)k0 * F_ + f;
      bf16* dt = at + tid * SK;
      bf16* dr = ar + tid * SK;
      float xc = src[0];
      #pragma unroll
      for (int i = 0; i < KT; i += 8) {
        bf16x8 vt, vr;
        #pragma unroll
        for (int j = 0; j < 8; ++j) {
          const int t  = k0 + i + j;
          const float xn = (t < T_ - 1) ? src[(i + j + 1) * F_] : xc;
          const float tr = 0.5f * (xc + xn);
          vt[j] = (bf16)tr;
          vr[j] = (bf16)(xc - tr);   // == 0 at t = T-1, matching reference
          xc = xn;
        }
        *(bf16x8*)(dt + i) = vt;     // ds_store_b128, K-contiguous
        *(bf16x8*)(dr + i) = vr;
      }
    }

    // W: coalesced float4 loads of two adjacent t-rows, transpose to [n][k]
    // with packed (k,k+1) 32-bit LDS stores.
    {
      const int kk = (tid >> 3) << 1;     // even t offset in chunk  (0..62)
      const int nc = (tid & 7) << 3;      // n offset in tile        (0..56)
      const size_t row0 = ((size_t)f * T_ + k0 + kk) * N_ + n0 + nc;
      const float* t0 = TW + row0;
      const float* r0 = RW + row0;
      float ta[8], tb[8], ra[8], rb[8];
      *(f32x4*)&ta[0] = *(const f32x4*)(t0);
      *(f32x4*)&ta[4] = *(const f32x4*)(t0 + 4);
      *(f32x4*)&tb[0] = *(const f32x4*)(t0 + N_);
      *(f32x4*)&tb[4] = *(const f32x4*)(t0 + N_ + 4);
      *(f32x4*)&ra[0] = *(const f32x4*)(r0);
      *(f32x4*)&ra[4] = *(const f32x4*)(r0 + 4);
      *(f32x4*)&rb[0] = *(const f32x4*)(r0 + N_);
      *(f32x4*)&rb[4] = *(const f32x4*)(r0 + N_ + 4);
      #pragma unroll
      for (int j = 0; j < 8; ++j) {
        bf16x2 pt; pt[0] = (bf16)ta[j]; pt[1] = (bf16)tb[j];
        bf16x2 pr; pr[0] = (bf16)ra[j]; pr[1] = (bf16)rb[j];
        *(bf16x2*)(wt + (nc + j) * SK + kk) = pt;  // ds_store_b32 (pairs)
        *(bf16x2*)(wr + (nc + j) * SK + kk) = pr;
      }
    }
  };

  // 16-bit WMMA operand fragment from K-contiguous LDS row:
  // lane<16 -> K [ks..ks+7] + [ks+16..ks+23]; lane>=16 -> +8 within each half.
  auto frag16 = [&](const bf16* base, int row, int ks) -> bf16x16 {
    const bf16* p = base + row * SK + ks + hs * 8;
    bf16x8 lo = *(const bf16x8*)(p);        // ds_load_b128
    bf16x8 hi = *(const bf16x8*)(p + 16);   // ds_load_b128
    return __builtin_shufflevector(lo, hi, 0,1,2,3,4,5,6,7,8,9,10,11,12,13,14,15);
  };

  load_stage(0, 0);
  __syncthreads();

  for (int kc = 0; kc < NC; ++kc) {
    const int s = kc & 1;
    if (kc + 1 < NC) load_stage(kc + 1, s ^ 1);   // prefetch next stage

    const bf16* at = sAt + s * A_STAGE;
    const bf16* ar = sAr + s * A_STAGE;
    const bf16* wt = sWt + s * W_STAGE;
    const bf16* wr = sWr + s * W_STAGE;

    #pragma unroll
    for (int ks = 0; ks < KT; ks += 32) {
      // Batch ALL fragment loads for this K-step into distinct registers
      // (12 frags x 8 VGPRs = 96 VGPRs) so the DS latency is covered by a
      // graduated s_wait_dscnt instead of a wait-0 before every WMMA, and
      // the 16 matrix ops issue as one dense XDL run.
      bf16x16 bt[2], br[2], atf[4], arf[4];
      #pragma unroll
      for (int nt = 0; nt < 2; ++nt) {
        const int col = wn * 32 + nt * 16 + l16;
        bt[nt] = frag16(wt, col, ks);
        br[nt] = frag16(wr, col, ks);
      }
      #pragma unroll
      for (int mt = 0; mt < 4; ++mt) {
        const int row = wm * 64 + mt * 16 + l16;
        atf[mt] = frag16(at, row, ks);
        arf[mt] = frag16(ar, row, ks);
      }
      #pragma unroll
      for (int mt = 0; mt < 4; ++mt) {
        #pragma unroll
        for (int nt = 0; nt < 2; ++nt) {
          acc[mt][nt] = __builtin_amdgcn_wmma_f32_16x16x32_bf16(
              false, atf[mt], false, bt[nt], (short)0, acc[mt][nt], false, false);
          acc[mt][nt] = __builtin_amdgcn_wmma_f32_16x16x32_bf16(
              false, arf[mt], false, br[nt], (short)0, acc[mt][nt], false, false);
        }
      }
    }
    __syncthreads();
  }

  // -------- epilogue: add fused biases, scatter to [B,N,F] layout ----------
  #pragma unroll
  for (int nt = 0; nt < 2; ++nt) {
    const int n = n0 + wn * 32 + nt * 16 + l16;
    const float bias = Tb[(size_t)f * N_ + n] + Rb[(size_t)f * N_ + n];
    #pragma unroll
    for (int mt = 0; mt < 4; ++mt) {
      #pragma unroll
      for (int j = 0; j < 8; ++j) {
        const int b = wm * 64 + mt * 16 + hs * 8 + j;  // C/D VGPR row mapping
        out[((size_t)b * N_ + n) * F_ + f] = acc[mt][nt][j] + bias;
      }
    }
  }
}

extern "C" void kernel_launch(void* const* d_in, const int* in_sizes, int n_in,
                              void* d_out, int out_size, void* d_ws, size_t ws_size,
                              hipStream_t stream) {
  const float* hist = (const float*)d_in[0];   // history_in [B,T,F]
  // d_in[1..7]: node_id / time_of_day / day_in_week / wt1..wt4 — unused by reference math
  const float* TW = (const float*)d_in[8];     // trend_W    [F,T,N]
  const float* Tb = (const float*)d_in[9];     // trend_b    [F,N]
  const float* RW = (const float*)d_in[10];    // residual_W [F,T,N]
  const float* Rb = (const float*)d_in[11];    // residual_b [F,N]
  float* out = (float*)d_out;                  // [B,N,F]

  (void)in_sizes; (void)n_in; (void)out_size; (void)d_ws; (void)ws_size;

  // ~180 KB dynamic LDS per workgroup (CDNA5 WGP has 320 KB) — opt in.
  hipFuncSetAttribute((const void*)dlinear_wmma_kernel,
                      hipFuncAttributeMaxDynamicSharedMemorySize,
                      (int)SMEM_BYTES);

  dim3 grid(N_ / NT, F_);   // 16 n-tiles x 16 features = 256 workgroups
  dim3 block(256);          // 8 wave32 per workgroup
  dlinear_wmma_kernel<<<grid, block, SMEM_BYTES, stream>>>(hist, TW, Tb, RW, Rb, out);
}
